// Model_39676907882670
// MI455X (gfx1250) — matrix operand order repaired and checked by reading, not compiled, not run on hardware
//
#include <hip/hip_runtime.h>

typedef __attribute__((ext_vector_type(16))) _Float16 v16h;
typedef __attribute__((ext_vector_type(8)))  _Float16 v8h;
typedef __attribute__((ext_vector_type(2)))  _Float16 v2h;
typedef __attribute__((ext_vector_type(8)))  float    v8f;

union HV16 { v16h v; v8h h[2]; };
union PK   { v2h h2; int i; };

#define S_LEN   2048
#define HEAD_D  64
#define BH_NUM  64      // B*H = 4*16
#define BQ      64      // q rows per workgroup (4 waves x 16)
#define BK      64      // keys per k-block iteration
#define LSTR    72      // LDS row stride in halves (64 + 8 pad -> conflict-free b128)
#define OSTR    68      // epilogue f32 row stride

// cross-half (lane ^ 16) exchange: VALU v_permlanex16_b32 with identity selectors
__device__ __forceinline__ int xor16_i32(int v) {
#if __has_builtin(__builtin_amdgcn_permlanex16)
  return __builtin_amdgcn_permlanex16(v, v, 0x76543210, (int)0xFEDCBA98, false, false);
#else
  return __shfl_xor(v, 16, 32);
#endif
}
__device__ __forceinline__ float xor16_f32(float v) {
  return __builtin_bit_cast(float, xor16_i32(__builtin_bit_cast(int, v)));
}

__global__ __launch_bounds__(128)
void fattn_f16wmma(const float* __restrict__ Q, const float* __restrict__ K,
                   const float* __restrict__ V, float* __restrict__ Out) {
  const int tid  = threadIdx.x;
  const int wave = tid >> 5;
  const int lane = tid & 31;
  const int lh   = lane & 15;   // N index (query) in all fragments
  const int hi   = lane >> 4;   // lane half-group

  const int bh = blockIdx.y;
  const int q0 = blockIdx.x * BQ;
  const size_t base = (size_t)bh * S_LEN * HEAD_D;

  // ldsK (9216 B) + ldsVt (9216 B) staged f16; epilogue reuses as 4x 16x68 f32
  __shared__ alignas(16) unsigned char smem[18432];
  _Float16* ldsK  = (_Float16*)smem;
  _Float16* ldsVt = (_Float16*)(smem + BK * LSTR * sizeof(_Float16));

  // ---- Q^T B-fragments: element e -> d = c*32 + hi*16 + e, N(q) = lane&15 ----
  HV16 qb[2];
  {
    const float* qp = Q + base + (size_t)(q0 + wave * 16 + lh) * HEAD_D;
    #pragma unroll
    for (int c = 0; c < 2; ++c) {
      const float* p = qp + c * 32 + hi * 16;
      #pragma unroll
      for (int g = 0; g < 2; ++g) {
        float4 f0 = *(const float4*)(p + g * 8);
        float4 f1 = *(const float4*)(p + g * 8 + 4);
        v8h h;
        h[0] = (_Float16)(f0.x * 0.125f); h[1] = (_Float16)(f0.y * 0.125f);
        h[2] = (_Float16)(f0.z * 0.125f); h[3] = (_Float16)(f0.w * 0.125f);
        h[4] = (_Float16)(f1.x * 0.125f); h[5] = (_Float16)(f1.y * 0.125f);
        h[6] = (_Float16)(f1.z * 0.125f); h[7] = (_Float16)(f1.w * 0.125f);
        qb[c].h[g] = h;
      }
    }
  }

  const v8f vzero = {0.f,0.f,0.f,0.f,0.f,0.f,0.f,0.f};
  v8f oacc[4];                       // O^T: element r -> d = ot*16 + hi*8 + r, q = lh
  #pragma unroll
  for (int o = 0; o < 4; ++o) oacc[o] = vzero;
  float m_ = -__builtin_inff(), l_ = 0.f;   // per-lane scalars (one q row per lane pair)

  const int row = tid >> 1;          // 0..63 : K staging row
  const int seg = (tid & 1) * 32;    // 0 or 32 : K staging column segment
  const int vg  = tid >> 4;          // 0..7  : V staging key-group (8 keys)
  const int vs  = tid & 15;          // d0 = vs*4 : V staging d-slice

  for (int kb = 0; kb < S_LEN / BK; ++kb) {
    __syncthreads();                 // protect LDS reuse across iterations
    // ---- stage K (row-major f16) ----
    {
      const float* kp = K + base + (size_t)(kb * BK + row) * HEAD_D + seg;
      #pragma unroll
      for (int j = 0; j < 32; j += 8) {
        float4 a = *(const float4*)(kp + j);
        float4 b = *(const float4*)(kp + j + 4);
        v8h h;
        h[0]=(_Float16)a.x; h[1]=(_Float16)a.y; h[2]=(_Float16)a.z; h[3]=(_Float16)a.w;
        h[4]=(_Float16)b.x; h[5]=(_Float16)b.y; h[6]=(_Float16)b.z; h[7]=(_Float16)b.w;
        *(v8h*)&ldsK[row * LSTR + seg + j] = h;
      }
    }
    // ---- stage V^T: thread owns 4(d) x 8(key) -> 4x ds_store_b128 ----
    {
      const float* vp = V + base + (size_t)(kb * BK + vg * 8) * HEAD_D + vs * 4;
      _Float16 hb[4][8];
      #pragma unroll
      for (int j = 0; j < 8; ++j) {
        float4 f = *(const float4*)(vp + j * HEAD_D);
        hb[0][j] = (_Float16)f.x; hb[1][j] = (_Float16)f.y;
        hb[2][j] = (_Float16)f.z; hb[3][j] = (_Float16)f.w;
      }
      #pragma unroll
      for (int dd = 0; dd < 4; ++dd) {
        v8h h;
        h[0]=hb[dd][0]; h[1]=hb[dd][1]; h[2]=hb[dd][2]; h[3]=hb[dd][3];
        h[4]=hb[dd][4]; h[5]=hb[dd][5]; h[6]=hb[dd][6]; h[7]=hb[dd][7];
        *(v8h*)&ldsVt[(vs * 4 + dd) * LSTR + vg * 8] = h;
      }
    }
    if (kb + 1 < S_LEN / BK) {       // overlap next block's HBM fetch
      __builtin_prefetch(K + base + (size_t)((kb + 1) * BK + row) * HEAD_D + seg, 0, 3);
      __builtin_prefetch(V + base + (size_t)((kb + 1) * BK + row) * HEAD_D + seg, 0, 3);
    }
    __syncthreads();

    // ---- S^T = K (Q/8)^T : A = K rows, B = Q^T. acc: M = key = hi*8+r, N = q ----
    v8f sacc[4];
    #pragma unroll
    for (int kt = 0; kt < 4; ++kt) {
      v8f acc = vzero;
      #pragma unroll
      for (int c = 0; c < 2; ++c) {
        const _Float16* kr = &ldsK[(kt * 16 + lh) * LSTR + c * 32 + hi * 16];
        HV16 ka;                        // A layout: e -> d = c*32+hi*16+e
        ka.h[0] = *(const v8h*)(kr);
        ka.h[1] = *(const v8h*)(kr + 8);
        acc = __builtin_amdgcn_wmma_f32_16x16x32_f16(false, ka.v, false, qb[c].v,
                                                     (short)0, acc, false, false);
      }
      sacc[kt] = acc;
    }

    // ---- online softmax: per-lane scalar state, one cross-half exchange each ----
    float mx = -__builtin_inff();
    #pragma unroll
    for (int kt = 0; kt < 4; ++kt)
      #pragma unroll
      for (int r = 0; r < 8; ++r) mx = fmaxf(mx, sacc[kt][r]);
    mx = fmaxf(mx, xor16_f32(mx));
    float mn = fmaxf(m_, mx);
    float corr = __expf(m_ - mn);
    m_ = mn;
    v8h ph[4];                        // exp'd scores as f16, per key-tile
    float rsum = 0.f;
    #pragma unroll
    for (int kt = 0; kt < 4; ++kt)
      #pragma unroll
      for (int r = 0; r < 8; ++r) {
        float p = __expf(sacc[kt][r] - mn);
        rsum += p;
        ph[kt][r] = (_Float16)p;
      }
    rsum += xor16_f32(rsum);
    l_ = l_ * corr + rsum;
    #pragma unroll
    for (int o = 0; o < 4; ++o)
      #pragma unroll
      for (int r = 0; r < 8; ++r) oacc[o][r] *= corr;   // rescale running O^T

    // ---- O^T += V^T P^T : build P^T B-frags straight from registers ----
    // pb[c] element e -> key = c*32 + hi*16 + e, N = q = lane&15
    HV16 pb[2];
    #pragma unroll
    for (int c = 0; c < 2; ++c) {
      #pragma unroll
      for (int j = 0; j < 8; ++j) {
        PK own; own.h2[0] = ph[2*c][j]; own.h2[1] = ph[2*c+1][j];
        PK rec; rec.i = xor16_i32(own.i);
        pb[c].v[j]     = hi ? rec.h2[1] : own.h2[0];
        pb[c].v[8 + j] = hi ? own.h2[1] : rec.h2[0];
      }
    }
    #pragma unroll
    for (int ot = 0; ot < 4; ++ot) {
      #pragma unroll
      for (int c = 0; c < 2; ++c) {
        const _Float16* vr = &ldsVt[(ot * 16 + lh) * LSTR + c * 32 + hi * 16];
        HV16 va;                        // A layout over V^T rows (M = d)
        va.h[0] = *(const v8h*)(vr);
        va.h[1] = *(const v8h*)(vr + 8);
        oacc[ot] = __builtin_amdgcn_wmma_f32_16x16x32_f16(false, va.v, false, pb[c].v,
                                                          (short)0, oacc[ot], false, false);
      }
    }
  }

  // ---- epilogue: normalize, transpose O^T -> O through LDS, coalesced stores ----
  __syncthreads();                    // all waves done reading ldsK/ldsVt
  {
    float inv = 1.f / l_;
    float* wO = (float*)smem + wave * (16 * OSTR);   // [q][d] per wave
    #pragma unroll
    for (int ot = 0; ot < 4; ++ot)
      #pragma unroll
      for (int r = 0; r < 8; r += 2) {
        float2 f2;
        f2.x = oacc[ot][r]     * inv;
        f2.y = oacc[ot][r + 1] * inv;
        *(float2*)&wO[lh * OSTR + ot * 16 + hi * 8 + r] = f2;
      }
    // wave-local: DS is in-order within a wave
    float* op = Out + base + (size_t)(q0 + wave * 16 + lh) * HEAD_D + hi * 32;
    #pragma unroll
    for (int j = 0; j < 8; ++j) {
      float4 f = *(const float4*)&wO[lh * OSTR + hi * 32 + j * 4];
      *(float4*)(op + j * 4) = f;
    }
  }
}

extern "C" void kernel_launch(void* const* d_in, const int* in_sizes, int n_in,
                              void* d_out, int out_size, void* d_ws, size_t ws_size,
                              hipStream_t stream) {
  const float* Q = (const float*)d_in[0];
  const float* K = (const float*)d_in[1];
  const float* V = (const float*)d_in[2];
  float* O = (float*)d_out;
  dim3 grid(S_LEN / BQ, BH_NUM);
  fattn_f16wmma<<<grid, dim3(128), 0, stream>>>(Q, K, V, O);
  (void)in_sizes; (void)n_in; (void)out_size; (void)d_ws; (void)ws_size;
}
